// InvariantPointAttention_5136780886393
// MI455X (gfx1250) — compile-verified, matrix operand order checked by reading, not run on hardware
//
#include <hip/hip_runtime.h>

// ---------------------------------------------------------------------------
// Invariant Point Attention for MI455X (gfx1250), bf16 WMMA path.
// ---------------------------------------------------------------------------

typedef __attribute__((ext_vector_type(16))) __bf16 v16bf;
typedef __attribute__((ext_vector_type(8)))  __bf16 v8bf;
typedef __attribute__((ext_vector_type(8)))  float  v8f;

#define IPA_B  2
#define IPA_N  768
#define IPA_H  12
#define IPA_C  16
#define IPA_PQ 4
#define IPA_PV 8
#define IPA_CN 384
#define IPA_T  (IPA_B * IPA_N)   // 1536 tokens
#define LDP    772               // LDS row stride (768 + 4) -> conflict-free

__device__ __forceinline__ __bf16 f2bf(float x) { return (__bf16)x; }

// Load one 16-bit WMMA operand fragment (A-layout == B-layout on transposed
// operand). Per ISA 7.12.2: lane holds row M=lane%16, kbase=(lane/16)*8,
// frag elems [0..7] = K kbase..kbase+7, [8..15] = K kbase+16..kbase+23.
// Caller passes p = &Mat[row*ld + k0 + kbase]; requires 16B alignment.
__device__ __forceinline__ v16bf load_frag16(const __bf16* p) {
  v8bf a = *(const v8bf*)p;
  v8bf b = *(const v8bf*)(p + 16);
  v16bf r;
#pragma unroll
  for (int i = 0; i < 8; ++i) { r[i] = a[i]; r[i + 8] = b[i]; }
  return r;
}

__device__ __forceinline__ v8f wmma_bf16(v16bf a, v16bf b, v8f c) {
  return __builtin_amdgcn_wmma_f32_16x16x32_bf16(
      /*neg_a=*/false, a, /*neg_b=*/false, b,
      /*c_mod=*/(short)0, c, /*reuse_a=*/false, /*reuse_b=*/false);
}

// ---------------------------------------------------------------------------
// f32 -> bf16 conversion
// ---------------------------------------------------------------------------
__global__ void cvt_f32_bf16(const float* __restrict__ src,
                             __bf16* __restrict__ dst, int n) {
  int i = blockIdx.x * blockDim.x + threadIdx.x;
  if (i < n) dst[i] = f2bf(src[i]);
}

// ---------------------------------------------------------------------------
// Generic bf16 GEMM: C[M x Nout] = A[M x K] * Bw[Nout x K]^T + bias[Nout]
// One wave (32 threads) per 16x16 C tile. K multiple of 32, M/Nout of 16.
// ---------------------------------------------------------------------------
__global__ __launch_bounds__(32) void gemm_bf16_kernel(
    const __bf16* __restrict__ A, const __bf16* __restrict__ Bw,
    const float* __restrict__ bias, float* __restrict__ C,
    int M, int Nout, int K) {
  int mi = blockIdx.x, ni = blockIdx.y;
  int l  = threadIdx.x;
  int lr = l & 15, lh = l >> 4;

  const __bf16* Ar = A  + (size_t)(mi * 16 + lr) * K + lh * 8;
  const __bf16* Br = Bw + (size_t)(ni * 16 + lr) * K + lh * 8;

  v8f acc = {};
  for (int k0 = 0; k0 < K; k0 += 32) {
    v16bf af = load_frag16(Ar + k0);
    v16bf bf = load_frag16(Br + k0);
    acc = wmma_bf16(af, bf, acc);
  }

  int col  = ni * 16 + lr;
  float bb = bias ? bias[col] : 0.0f;
#pragma unroll
  for (int r = 0; r < 8; ++r) {
    int row = mi * 16 + r + 8 * lh;   // ISA C/D layout
    C[(size_t)row * Nout + col] = acc[r] + bb;
  }
}

// ---------------------------------------------------------------------------
// Per-token geometry + operand packing.
// Builds (per b,h):
//   Qaug[n][0:16]=q*s1, [16:28]=hw*q_pts, [28:32]=0        (bf16, [BH][N][32])
//   Kaug[n][0:16]=k,    [16:28]=k_pts,    [28:32]=0        (bf16, [BH][N][32])
//   Vt[feat][n]: feat 0:16=v, 16:40=v_pts(p,x), 40:48=0    (bf16, [BH][48][N])
//   biasQ[n] = -0.5*hw*sum_p|q_pts|^2,  biasK analogous    (f32,  [BH][N])
// ---------------------------------------------------------------------------
__global__ void ipa_transform(
    const float* __restrict__ qbuf,    // [T][192]
    const float* __restrict__ kvbuf,   // [T][384]
    const float* __restrict__ qpraw,   // [T][144]
    const float* __restrict__ kvpraw,  // [T][432]
    const float* __restrict__ rot,     // [T][9]
    const float* __restrict__ trans,   // [T][3]
    const float* __restrict__ hweights,// [H]
    __bf16* __restrict__ Qaug, __bf16* __restrict__ Kaug,
    __bf16* __restrict__ Vt,
    float* __restrict__ biasQ, float* __restrict__ biasK) {
  int t = blockIdx.x * blockDim.x + threadIdx.x;
  if (t >= IPA_T) return;
  int b = t / IPA_N, n = t % IPA_N;

  float R[9], tr[3];
#pragma unroll
  for (int i = 0; i < 9; ++i) R[i] = rot[(size_t)t * 9 + i];
#pragma unroll
  for (int i = 0; i < 3; ++i) tr[i] = trans[(size_t)t * 3 + i];

  const float s1       = 0.14433756729740643f;  // 1/sqrt(3*C) = 1/sqrt(48)
  const float hw_scale = 0.13608276348795434f;  // sqrt(1/(3*PQ*9/2)) = 1/sqrt(54)

  for (int h = 0; h < IPA_H; ++h) {
    float hw = log1pf(__expf(hweights[h])) * hw_scale;  // softplus * scale
    size_t bh = (size_t)b * IPA_H + h;
    __bf16* Qrow = Qaug + (bh * IPA_N + n) * 32;
    __bf16* Krow = Kaug + (bh * IPA_N + n) * 32;
    __bf16* Vb   = Vt   + bh * 48 * IPA_N;

#pragma unroll
    for (int c = 0; c < 16; ++c) {
      Qrow[c] = f2bf(qbuf[(size_t)t * 192 + h * 16 + c] * s1);
      Krow[c] = f2bf(kvbuf[(size_t)t * 384 + h * 32 + c]);
      Vb[(size_t)c * IPA_N + n] = f2bf(kvbuf[(size_t)t * 384 + h * 32 + 16 + c]);
    }

    // q points: raw[t, x*48 + h*PQ + p] -> rotate + translate
    float qn2 = 0.0f;
#pragma unroll
    for (int p = 0; p < IPA_PQ; ++p) {
      float px[3], v[3];
#pragma unroll
      for (int x = 0; x < 3; ++x)
        px[x] = qpraw[(size_t)t * 144 + x * 48 + h * IPA_PQ + p];
#pragma unroll
      for (int x = 0; x < 3; ++x) {
        v[x] = R[x * 3 + 0] * px[0] + R[x * 3 + 1] * px[1] + R[x * 3 + 2] * px[2] + tr[x];
        qn2 += v[x] * v[x];
        Qrow[16 + p * 3 + x] = f2bf(hw * v[x]);
      }
    }
#pragma unroll
    for (int c = 28; c < 32; ++c) { Qrow[c] = f2bf(0.0f); Krow[c] = f2bf(0.0f); }

    // kv points: raw[t, x*144 + h*(PQ+PV) + pp]
    float kn2 = 0.0f;
#pragma unroll
    for (int pp = 0; pp < IPA_PQ + IPA_PV; ++pp) {
      float px[3], v[3];
#pragma unroll
      for (int x = 0; x < 3; ++x)
        px[x] = kvpraw[(size_t)t * 432 + x * 144 + h * (IPA_PQ + IPA_PV) + pp];
#pragma unroll
      for (int x = 0; x < 3; ++x) {
        v[x] = R[x * 3 + 0] * px[0] + R[x * 3 + 1] * px[1] + R[x * 3 + 2] * px[2] + tr[x];
        if (pp < IPA_PQ) {
          kn2 += v[x] * v[x];
          Krow[16 + pp * 3 + x] = f2bf(v[x]);
        } else {
          Vb[(size_t)(16 + (pp - IPA_PQ) * 3 + x) * IPA_N + n] = f2bf(v[x]);
        }
      }
    }
#pragma unroll
    for (int f = 40; f < 48; ++f) Vb[(size_t)f * IPA_N + n] = f2bf(0.0f);

    biasQ[bh * IPA_N + n] = -0.5f * hw * qn2;
    biasK[bh * IPA_N + n] = -0.5f * hw * kn2;
  }
}

// ---------------------------------------------------------------------------
// Fused attention: one wave per (b, h, 16-row tile).
// Logits: single 16x16x32 WMMA per j-tile (K=28 padded to 32), + biases/mask.
// Softmax in LDS; 1/rowsum folded into bf16 A-fragment conversion.
// AV: 3 accumulators (48 feature cols) over K=768.
// ---------------------------------------------------------------------------
__global__ __launch_bounds__(32) void ipa_attention(
    const __bf16* __restrict__ Qaug, const __bf16* __restrict__ Kaug,
    const __bf16* __restrict__ Vt,
    const float* __restrict__ biasQ, const float* __restrict__ biasK,
    const float* __restrict__ pair_mask,   // [B][N][N]
    float* __restrict__ Obuf) {            // [B][N][H][48]
  __shared__ float Lds[16 * LDP];

  int ntiles = IPA_N / 16;
  int it = blockIdx.x % ntiles;
  int bh = blockIdx.x / ntiles;
  int b  = bh / IPA_H, h = bh % IPA_H;
  int l  = threadIdx.x, lr = l & 15, lh = l >> 4;
  int i0 = it * 16;

  const __bf16* Q  = Qaug + (size_t)bh * IPA_N * 32;
  const __bf16* Kp = Kaug + (size_t)bh * IPA_N * 32;
  const __bf16* V  = Vt   + (size_t)bh * 48 * IPA_N;
  const float*  bQ = biasQ + (size_t)bh * IPA_N;
  const float*  bK = biasK + (size_t)bh * IPA_N;
  const float*  Mk = pair_mask + ((size_t)b * IPA_N + i0) * IPA_N;

  // Q fragment for this row tile (K = 32, single fragment).
  v16bf qf = load_frag16(Q + (size_t)(i0 + lr) * 32 + lh * 8);

  // ---- logits into LDS ----
  for (int jt = 0; jt < ntiles; ++jt) {
    v16bf kf = load_frag16(Kp + (size_t)(jt * 16 + lr) * 32 + lh * 8);
    v8f c = {};
    c = wmma_bf16(qf, kf, c);
    int j = jt * 16 + lr;
    float bj = bK[j];
#pragma unroll
    for (int r = 0; r < 8; ++r) {
      int rl = r + 8 * lh;                       // local row (C/D layout)
      float val = c[r] + bQ[i0 + rl] + bj
                + 100000.0f * (Mk[(size_t)rl * IPA_N + j] - 1.0f);
      Lds[rl * LDP + jt * 16 + lr] = val;
    }
  }

  // ---- softmax: lanes (l, l+16) share row lr; each scans one half ----
  int jb = lh * (IPA_N / 2);
  float m = -3.4e38f;
  for (int j = 0; j < IPA_N / 2; ++j) m = fmaxf(m, Lds[lr * LDP + jb + j]);
  m = fmaxf(m, __shfl_xor(m, 16, 32));
  float ssum = 0.0f;
  for (int j = 0; j < IPA_N / 2; ++j) {
    float e = __expf(Lds[lr * LDP + jb + j] - m);
    Lds[lr * LDP + jb + j] = e;
    ssum += e;
  }
  ssum += __shfl_xor(ssum, 16, 32);
  float rinv = 1.0f / ssum;   // lane's fragment row is lr -> fold into A-frag

  // ---- O = P * [V | V_pts] (feature width 48 = 3 N-tiles) ----
  v8f acc0 = {}, acc1 = {}, acc2 = {};
  for (int k0 = 0; k0 < IPA_N; k0 += 32) {
    v16bf af;
    const float* Lr = Lds + lr * LDP + k0 + lh * 8;
#pragma unroll
    for (int i = 0; i < 8; ++i) {
      af[i]     = f2bf(Lr[i] * rinv);
      af[i + 8] = f2bf(Lr[i + 16] * rinv);
    }
    v16bf b0 = load_frag16(V + (size_t)(0  + lr) * IPA_N + k0 + lh * 8);
    acc0 = wmma_bf16(af, b0, acc0);
    v16bf b1 = load_frag16(V + (size_t)(16 + lr) * IPA_N + k0 + lh * 8);
    acc1 = wmma_bf16(af, b1, acc1);
    v16bf b2 = load_frag16(V + (size_t)(32 + lr) * IPA_N + k0 + lh * 8);
    acc2 = wmma_bf16(af, b2, acc2);
  }

#pragma unroll
  for (int r = 0; r < 8; ++r) {
    int i = i0 + r + 8 * lh;
    float* Op = Obuf + (((size_t)b * IPA_N + i) * IPA_H + h) * 48;
    Op[lr]      = acc0[r];
    Op[16 + lr] = acc1[r];
    Op[32 + lr] = acc2[r];
  }
}

// ---------------------------------------------------------------------------
// Finisher: inverse rotation, norms, build concat vector (bf16, 576 wide).
// cat = [o(192) | o_pt.x(96) | o_pt.y(96) | o_pt.z(96) | norm(96)]
// ---------------------------------------------------------------------------
__global__ void ipa_finish(const float* __restrict__ Obuf,   // [T][H][48]
                           const float* __restrict__ rot,    // [T][9]
                           const float* __restrict__ trans,  // [T][3]
                           __bf16* __restrict__ catbf) {     // [T][576]
  int t = blockIdx.x * blockDim.x + threadIdx.x;
  if (t >= IPA_T) return;

  float R[9], tr[3];
#pragma unroll
  for (int i = 0; i < 9; ++i) R[i] = rot[(size_t)t * 9 + i];
#pragma unroll
  for (int i = 0; i < 3; ++i) tr[i] = trans[(size_t)t * 3 + i];

  __bf16* cat = catbf + (size_t)t * 576;
  for (int h = 0; h < IPA_H; ++h) {
    const float* Ob = Obuf + ((size_t)t * IPA_H + h) * 48;
#pragma unroll
    for (int c = 0; c < 16; ++c) cat[h * 16 + c] = f2bf(Ob[c]);
#pragma unroll
    for (int p = 0; p < IPA_PV; ++p) {
      float w0 = Ob[16 + p * 3 + 0] - tr[0];
      float w1 = Ob[16 + p * 3 + 1] - tr[1];
      float w2 = Ob[16 + p * 3 + 2] - tr[2];
      // e[x] = sum_y R[y][x] * w[y]   (rot^T)
      float e0 = R[0] * w0 + R[3] * w1 + R[6] * w2;
      float e1 = R[1] * w0 + R[4] * w1 + R[7] * w2;
      float e2 = R[2] * w0 + R[5] * w1 + R[8] * w2;
      float nrm = sqrtf(e0 * e0 + e1 * e1 + e2 * e2 + 1e-7f);
      cat[192 + h * IPA_PV + p] = f2bf(e0);
      cat[288 + h * IPA_PV + p] = f2bf(e1);
      cat[384 + h * IPA_PV + p] = f2bf(e2);
      cat[480 + h * IPA_PV + p] = f2bf(nrm);
    }
  }
}

// ---------------------------------------------------------------------------
// Host-side orchestration
// ---------------------------------------------------------------------------
static inline size_t alignup(size_t x) { return (x + 255) & ~(size_t)255; }

extern "C" void kernel_launch(void* const* d_in, const int* in_sizes, int n_in,
                              void* d_out, int out_size, void* d_ws, size_t ws_size,
                              hipStream_t stream) {
  (void)in_sizes; (void)n_in; (void)out_size; (void)ws_size;
  const float* s        = (const float*)d_in[0];
  const float* rot      = (const float*)d_in[1];
  const float* trans    = (const float*)d_in[2];
  const float* mask     = (const float*)d_in[3];
  const float* w_q      = (const float*)d_in[4];
  const float* b_q      = (const float*)d_in[5];
  const float* w_kv     = (const float*)d_in[6];
  const float* b_kv     = (const float*)d_in[7];
  const float* w_qp     = (const float*)d_in[8];
  const float* b_qp     = (const float*)d_in[9];
  const float* w_kvp    = (const float*)d_in[10];
  const float* b_kvp    = (const float*)d_in[11];
  const float* hweights = (const float*)d_in[12];
  const float* w_out    = (const float*)d_in[13];
  const float* b_out    = (const float*)d_in[14];
  float* out = (float*)d_out;

  const int T = IPA_T, BH = IPA_B * IPA_H;
  char* ws = (char*)d_ws;
  size_t off = 0;
  auto bump = [&](size_t bytes) { char* p = ws + off; off = alignup(off + bytes); return p; };

  __bf16* sbf    = (__bf16*)bump((size_t)T * 384 * 2);
  __bf16* wqbf   = (__bf16*)bump((size_t)192 * 384 * 2);
  __bf16* wkvbf  = (__bf16*)bump((size_t)384 * 384 * 2);
  __bf16* wqpbf  = (__bf16*)bump((size_t)144 * 384 * 2);
  __bf16* wkvpbf = (__bf16*)bump((size_t)432 * 384 * 2);
  __bf16* woutbf = (__bf16*)bump((size_t)384 * 576 * 2);
  float*  qbuf   = (float*)bump((size_t)T * 192 * 4);
  float*  kvbuf  = (float*)bump((size_t)T * 384 * 4);
  float*  qpraw  = (float*)bump((size_t)T * 144 * 4);
  float*  kvpraw = (float*)bump((size_t)T * 432 * 4);
  __bf16* Qaug   = (__bf16*)bump((size_t)BH * IPA_N * 32 * 2);
  __bf16* Kaug   = (__bf16*)bump((size_t)BH * IPA_N * 32 * 2);
  __bf16* Vt     = (__bf16*)bump((size_t)BH * 48 * IPA_N * 2);
  float*  biasQ  = (float*)bump((size_t)BH * IPA_N * 4);
  float*  biasK  = (float*)bump((size_t)BH * IPA_N * 4);
  float*  Obuf   = (float*)bump((size_t)T * IPA_H * 48 * 4);
  __bf16* catbf  = (__bf16*)bump((size_t)T * 576 * 2);

  auto cvt = [&](const float* src, __bf16* dst, int n) {
    cvt_f32_bf16<<<(n + 255) / 256, 256, 0, stream>>>(src, dst, n);
  };
  cvt(s, sbf, T * 384);
  cvt(w_q, wqbf, 192 * 384);
  cvt(w_kv, wkvbf, 384 * 384);
  cvt(w_qp, wqpbf, 144 * 384);
  cvt(w_kvp, wkvpbf, 432 * 384);
  cvt(w_out, woutbf, 384 * 576);

  auto gemm = [&](const __bf16* A, const __bf16* Bw, const float* bias,
                  float* C, int M, int Nout, int K) {
    dim3 grid(M / 16, Nout / 16);
    gemm_bf16_kernel<<<grid, 32, 0, stream>>>(A, Bw, bias, C, M, Nout, K);
  };
  gemm(sbf, wqbf,   b_q,   qbuf,   T, 192, 384);
  gemm(sbf, wkvbf,  b_kv,  kvbuf,  T, 384, 384);
  gemm(sbf, wqpbf,  b_qp,  qpraw,  T, 144, 384);
  gemm(sbf, wkvpbf, b_kvp, kvpraw, T, 432, 384);

  ipa_transform<<<(T + 127) / 128, 128, 0, stream>>>(
      qbuf, kvbuf, qpraw, kvpraw, rot, trans, hweights,
      Qaug, Kaug, Vt, biasQ, biasK);

  ipa_attention<<<BH * (IPA_N / 16), 32, 0, stream>>>(
      Qaug, Kaug, Vt, biasQ, biasK, mask, Obuf);

  ipa_finish<<<(T + 127) / 128, 128, 0, stream>>>(Obuf, rot, trans, catbf);

  gemm(catbf, woutbf, b_out, out, T, 384, 576);
}
